// PointNetSetAbstractionMsg_21182778704778
// MI455X (gfx1250) — compile-verified, matrix operand order checked by reading, not run on hardware
//
#include <hip/hip_runtime.h>
#include <hip/hip_bf16.h>

// PointNet++ MSG set abstraction for MI455X (gfx1250, wave32, WMMA).
// Activations stored f16; GEMM via v_wmma_f32_16x16x32_f16 (f32 accum).
// Each wave computes ALL Cout tiles for its 16-sample strip -> X read once,
// BN-affine+ReLU applied once per element (fused into the B-tile load).
// BN stats via a separate cheap reduction over the stored z (f16).
// Conv bias omitted: BN's mean subtraction cancels it exactly.

typedef _Float16 v16h __attribute__((ext_vector_type(16)));
typedef _Float16 v8h  __attribute__((ext_vector_type(8)));
typedef float    v8f  __attribute__((ext_vector_type(8)));

#define NPTS 4096
#define B_   16
#define S_   1024
#define EPS_ 1e-5f

__device__ inline v16h cat8(v8h lo, v8h hi) {
  v16h r;
#pragma unroll
  for (int i = 0; i < 8; ++i) { r[i] = lo[i]; r[8 + i] = hi[i]; }
  return r;
}

// ---------------- FPS: one 1024-thread workgroup per batch -----------------
__global__ __launch_bounds__(1024) void fps_kernel(
    const float* __restrict__ coord, int* __restrict__ fps_idx,
    float* __restrict__ coord_ce, float* __restrict__ coord_out)
{
  __shared__ float sx[NPTS], sy[NPTS], sz[NPTS];
  __shared__ float wbv[32];
  __shared__ int   wbi[32];
  __shared__ int   s_far;
  const int b = blockIdx.x, tid = threadIdx.x;
  const int lane = tid & 31, wv = tid >> 5;
  const float* cb = coord + (size_t)b * 3 * NPTS;
#pragma unroll
  for (int t = 0; t < 4; ++t) {
    int n = tid + t * 1024;
    sx[n] = cb[n]; sy[n] = cb[NPTS + n]; sz[n] = cb[2 * NPTS + n];
  }
  float dist[4] = {1e10f, 1e10f, 1e10f, 1e10f};
  int far = 0;
  __syncthreads();
  for (int s = 0; s < S_; ++s) {
    float cx = sx[far], cy = sy[far], cz = sz[far];
    if (tid == 0) {
      fps_idx[b * S_ + s] = far;
      float* ce = coord_ce + ((size_t)b * S_ + s) * 3;
      ce[0] = cx; ce[1] = cy; ce[2] = cz;
      coord_out[((size_t)b * 3 + 0) * S_ + s] = cx;
      coord_out[((size_t)b * 3 + 1) * S_ + s] = cy;
      coord_out[((size_t)b * 3 + 2) * S_ + s] = cz;
    }
    float bvv = -1.0f; int bii = 0;
#pragma unroll
    for (int t = 0; t < 4; ++t) {
      int n = tid + t * 1024;
      float dx = sx[n] - cx, dy = sy[n] - cy, dz = sz[n] - cz;
      float nd = fminf(dist[t], dx * dx + dy * dy + dz * dz);
      dist[t] = nd;
      if (nd > bvv || (nd == bvv && n < bii)) { bvv = nd; bii = n; }
    }
#pragma unroll
    for (int off = 16; off >= 1; off >>= 1) {   // wave32 butterfly argmax
      float ov = __shfl_xor(bvv, off, 32);
      int   oi = __shfl_xor(bii, off, 32);
      if (ov > bvv || (ov == bvv && oi < bii)) { bvv = ov; bii = oi; }
    }
    if (lane == 0) { wbv[wv] = bvv; wbi[wv] = bii; }
    __syncthreads();
    if (tid < 32) {
      bvv = wbv[tid]; bii = wbi[tid];
#pragma unroll
      for (int off = 16; off >= 1; off >>= 1) {
        float ov = __shfl_xor(bvv, off, 32);
        int   oi = __shfl_xor(bii, off, 32);
        if (ov > bvv || (ov == bvv && oi < bii)) { bvv = ov; bii = oi; }
      }
      if (tid == 0) s_far = bii;
    }
    __syncthreads();
    far = s_far;
  }
}

// ---------------- Ball query: one wave per (b, centroid) -------------------
__global__ __launch_bounds__(256) void ball_kernel(
    const float* __restrict__ coord, const float* __restrict__ coord_ce,
    int* __restrict__ idx, int K, float r2)
{
  int g = blockIdx.x * 8 + (threadIdx.x >> 5);
  int lane = threadIdx.x & 31;
  int b = g >> 10, s = g & 1023;
  const float* ce = coord_ce + ((size_t)b * S_ + s) * 3;
  float cx = ce[0], cy = ce[1], cz = ce[2];
  const float* cb = coord + (size_t)b * 3 * NPTS;
  int* out = idx + (size_t)g * K;
  int cnt = 0, firstIdx = -1;
  for (int base = 0; base < NPTS && cnt < K; base += 32) {
    int n = base + lane;
    float dx = cb[n] - cx, dy = cb[NPTS + n] - cy, dz = cb[2 * NPTS + n] - cz;
    bool pred = (dx * dx + dy * dy + dz * dz) <= r2;
    unsigned mask = (unsigned)__ballot((int)pred);
    if (firstIdx < 0 && mask) firstIdx = base + __ffs(mask) - 1;
    int prefix = __popc(mask & ((1u << lane) - 1u));
    if (pred && (cnt + prefix) < K) out[cnt + prefix] = n;
    cnt += __popc(mask);
  }
  if (cnt > K) cnt = K;
  if (firstIdx < 0) firstIdx = 0;
  for (int p = cnt + lane; p < K; p += 32) out[p] = firstIdx;  // pad w/ first
}

// ---------------- Gather: build x0 [M, 64] f16 (35 real ch + zero pad) -----
__global__ __launch_bounds__(256) void gather_kernel(
    const float* __restrict__ feat, const float* __restrict__ coord,
    const float* __restrict__ coord_ce, const int* __restrict__ idx,
    _Float16* __restrict__ X, int kLog2, long Mtot)
{
  long e = (long)blockIdx.x * 256 + threadIdx.x;
  long m = e >> 6;
  if (m >= Mtot) return;
  int ci = (int)(e & 63);
  long sb = m >> kLog2;
  int s = (int)(sb & 1023);
  int b = (int)(sb >> 10);
  int id = idx[m];
  float v;
  if (ci < 32)      v = feat[((long)b * 32 + ci) * NPTS + id];
  else if (ci < 35) {
    int d = ci - 32;
    v = coord[((long)b * 3 + d) * NPTS + id] - coord_ce[((long)b * S_ + s) * 3 + d];
  } else            v = 0.f;
  X[m * 64 + ci] = (_Float16)v;
}

// ---------------- Weight prep: f32 (Cout,Cin) -> f16 (Cout,CinPad) ---------
__global__ __launch_bounds__(256) void wprep_kernel(
    const float* __restrict__ Wsrc, _Float16* __restrict__ Wdst,
    int Cout, int Cin, int CinPad)
{
  int e = blockIdx.x * 256 + threadIdx.x;
  if (e >= Cout * CinPad) return;
  int co = e / CinPad, ci = e % CinPad;
  Wdst[e] = (_Float16)(ci < Cin ? Wsrc[co * Cin + ci] : 0.f);
}

__global__ void init_id(float* idA, float* idS) {
  int c = threadIdx.x; idA[c] = 1.f; idS[c] = 0.f;
}

// ---------------- GEMM with fused input BN-affine/ReLU ---------------------
// block = 8 waves x 16 samples = 128 M rows; each wave computes ALL Cout
// tiles (NCT = Cout/16) for its strip. Weights staged in LDS once per block.
template <int NCT>
__global__ __launch_bounds__(256) void gemm_kernel(
    const _Float16* __restrict__ X, const _Float16* __restrict__ Wf,
    const float* __restrict__ trA, const float* __restrict__ trS, int relu_in,
    _Float16* __restrict__ Z, int CinPad)
{
  constexpr int Cout = NCT * 16;
  __shared__ float s_a[128], s_s[128];
  __shared__ _Float16 sW[12288];           // up to 128 x 96 f16 weights
  const int tid = threadIdx.x;
  if (tid < CinPad) { s_a[tid] = trA[tid]; s_s[tid] = trS[tid]; }
  const int wtot = Cout * CinPad;          // multiple of 1024
  for (int e = tid * 8; e < wtot; e += 256 * 8)
    *(v8h*)(sW + e) = *(const v8h*)(Wf + e);
  __syncthreads();
  const int lane = tid & 31, wv = tid >> 5;
  const int hh = lane >> 4, r = lane & 15;
  const long m_row = (long)blockIdx.x * 128 + wv * 16 + r;
  v8f acc[NCT];
#pragma unroll
  for (int ct = 0; ct < NCT; ++ct) acc[ct] = v8f{};
  for (int ci = 0; ci < CinPad; ci += 32) {
    const int k0 = ci + hh * 8, k1 = ci + 16 + hh * 8;
    const _Float16* xp = X + m_row * (long)CinPad + k0;
    v8h b0 = *(const v8h*)xp;
    v8h b1 = *(const v8h*)(xp + 16);
    v16h bf;
#pragma unroll
    for (int j = 0; j < 8; ++j) {          // fused prev-layer BN affine + ReLU
      float y0 = fmaf(s_a[k0 + j], (float)b0[j], s_s[k0 + j]);
      float y1 = fmaf(s_a[k1 + j], (float)b1[j], s_s[k1 + j]);
      if (relu_in) { y0 = fmaxf(y0, 0.f); y1 = fmaxf(y1, 0.f); }
      bf[j] = (_Float16)y0; bf[8 + j] = (_Float16)y1;
    }
#pragma unroll
    for (int ct = 0; ct < NCT; ++ct) {
      const _Float16* wp = sW + (ct * 16 + r) * CinPad + k0;
      v8h a0 = *(const v8h*)wp;
      v8h a1 = *(const v8h*)(wp + 16);
      v16h af = cat8(a0, a1);
      acc[ct] = __builtin_amdgcn_wmma_f32_16x16x32_f16(
          false, af, false, bf, (short)0, acc[ct], false, false);
    }
  }
  _Float16* zrow = Z + m_row * (long)Cout + hh * 8;
#pragma unroll
  for (int ct = 0; ct < NCT; ++ct) {
    v8h outv;
#pragma unroll
    for (int j = 0; j < 8; ++j) outv[j] = (_Float16)acc[ct][j];
    *(v8h*)(zrow + ct * 16) = outv;
  }
}

// ---------------- per-channel sum / sumsq over z (f16) ---------------------
__global__ __launch_bounds__(256) void stats_kernel(
    const _Float16* __restrict__ Z, float* __restrict__ gsum,
    float* __restrict__ gsqs, int Cout, long M, int rowsPerBlock)
{
  __shared__ float s_sum[128], s_sqs[128];
  const int tid = threadIdx.x;
  if (tid < Cout) { s_sum[tid] = 0.f; s_sqs[tid] = 0.f; }
  __syncthreads();
  const int groups = 256 / Cout;           // 8 / 4 / 2 (Cout 32/64/96|128)
  const int active = groups * Cout;
  if (tid < active) {
    const int c = tid % Cout, g = tid / Cout;
    long r0 = (long)blockIdx.x * rowsPerBlock;
    long r1 = r0 + rowsPerBlock; if (r1 > M) r1 = M;
    float ls = 0.f, lq = 0.f;
    for (long rr = r0 + g; rr < r1; rr += groups) {
      float z = (float)Z[rr * Cout + c];
      ls += z; lq += z * z;
    }
    atomicAdd(&s_sum[c], ls);
    atomicAdd(&s_sqs[c], lq);
  }
  __syncthreads();
  if (tid < Cout) {
    atomicAdd(&gsum[tid], s_sum[tid]);
    atomicAdd(&gsqs[tid], s_sqs[tid]);
  }
}

// ---------------- BN stats -> affine (a, shift) ----------------------------
__global__ void finalize_kernel(const float* gsum, const float* gsqs,
                                const float* g, const float* beta,
                                float* trA, float* trS, float invM, int Cout)
{
  int c = threadIdx.x;
  if (c >= Cout) return;
  float mean = gsum[c] * invM;
  float var  = gsqs[c] * invM - mean * mean;
  float a    = g[c] * rsqrtf(var + EPS_);
  trA[c] = a;
  trS[c] = beta[c] - mean * a;
}

// ---------------- affine+ReLU fused max over K -----------------------------
__global__ __launch_bounds__(256) void maxk_kernel(
    const _Float16* __restrict__ Z, const float* __restrict__ trA,
    const float* __restrict__ trS, float* __restrict__ outFeat,
    int K, int C, int c_off)
{
  long e = (long)blockIdx.x * 256 + threadIdx.x;
  long total = (long)B_ * S_ * C;
  if (e >= total) return;
  int c = (int)(e % C);
  long bs = e / C;
  int s = (int)(bs & 1023);
  int b = (int)(bs >> 10);
  float a = trA[c], sh = trS[c];
  const _Float16* zp = Z + bs * (long)K * C + c;
  float m = 0.f;  // ReLU output >= 0
  for (int j = 0; j < K; ++j)
    m = fmaxf(m, fmaxf(fmaf(a, (float)zp[(long)j * C], sh), 0.f));
  outFeat[((long)b * 320 + c_off + c) * S_ + s] = m;
}

static void launch_gemm(int nct, const _Float16* X, const _Float16* Wf,
                        const float* trA, const float* trS, int relu_in,
                        _Float16* Z, int CinPad, long M, hipStream_t stream)
{
  dim3 grid((unsigned)(M / 128));
  switch (nct) {
    case 2: gemm_kernel<2><<<grid, 256, 0, stream>>>(X, Wf, trA, trS, relu_in, Z, CinPad); break;
    case 4: gemm_kernel<4><<<grid, 256, 0, stream>>>(X, Wf, trA, trS, relu_in, Z, CinPad); break;
    case 6: gemm_kernel<6><<<grid, 256, 0, stream>>>(X, Wf, trA, trS, relu_in, Z, CinPad); break;
    default: gemm_kernel<8><<<grid, 256, 0, stream>>>(X, Wf, trA, trS, relu_in, Z, CinPad); break;
  }
}

extern "C" void kernel_launch(void* const* d_in, const int* in_sizes, int n_in,
                              void* d_out, int out_size, void* d_ws, size_t ws_size,
                              hipStream_t stream)
{
  const float* feat  = (const float*)d_in[0];
  const float* coord = (const float*)d_in[1];
  // params flattened per JAX pytree order: per block, per layer: W, b, beta, g
  const float *Wp[9], *betap[9], *gp[9];
  for (int l = 0; l < 9; ++l) {
    int base = 2 + l * 4;
    Wp[l]    = (const float*)d_in[base + 0];
    // d_in[base+1] is the conv bias: mathematically cancelled by BN mean-sub.
    betap[l] = (const float*)d_in[base + 2];
    gp[l]    = (const float*)d_in[base + 3];
  }
  char* ws = (char*)d_ws;
  int*      fpsI  = (int*)(ws + 0);               // 64 KB
  float*    ce    = (float*)(ws + 65536);         // 192 KB
  int*      ballI = (int*)(ws + 262144);          // 8 MB (reused per scale)
  float*    gsum  = (float*)(ws + 8650752);
  float*    gsqs  = (float*)(ws + 8651264);
  float*    trA   = (float*)(ws + 8651776);
  float*    trS   = (float*)(ws + 8652288);
  float*    idA   = (float*)(ws + 8652800);
  float*    idS   = (float*)(ws + 8653312);
  _Float16* Wh    = (_Float16*)(ws + 8653824);    // 9 layers * 32 KB
  _Float16* bufA  = (_Float16*)(ws + 8948736);    // M_max * 96 f16  (403 MB)
  _Float16* bufB  = (_Float16*)(ws + 411601920);  // M_max * 128 f16 (537 MB)

  static const int CIN[9]  = {35, 32, 32, 35, 64, 64, 35, 64, 96};
  static const int CINP[9] = {64, 32, 32, 64, 64, 64, 64, 64, 96};
  static const int COUT[9] = {32, 32, 64, 64, 64, 128, 64, 96, 128};
  static const int KS[3]   = {16, 32, 128};
  static const int KLOG[3] = {4, 5, 7};
  static const float RS[3] = {0.1f, 0.2f, 0.4f};
  static const int COFF[3] = {0, 64, 192};

  float* outFeat  = (float*)d_out;
  float* outCoord = outFeat + (size_t)B_ * 320 * S_;

  init_id<<<1, 128, 0, stream>>>(idA, idS);
  for (int l = 0; l < 9; ++l) {
    int tot = COUT[l] * CINP[l];
    wprep_kernel<<<(tot + 255) / 256, 256, 0, stream>>>(Wp[l], Wh + l * 16384,
                                                        COUT[l], CIN[l], CINP[l]);
  }
  fps_kernel<<<B_, 1024, 0, stream>>>(coord, fpsI, ce, outCoord);

  for (int sc = 0; sc < 3; ++sc) {
    int K = KS[sc];
    long M = (long)B_ * S_ * K;
    ball_kernel<<<(B_ * S_) / 8, 256, 0, stream>>>(coord, ce, ballI, K,
                                                   RS[sc] * RS[sc]);
    long e0 = M * 64;
    gather_kernel<<<(unsigned)((e0 + 255) / 256), 256, 0, stream>>>(
        feat, coord, ce, ballI, bufA, KLOG[sc], M);
    _Float16* xin = bufA;
    _Float16* xout = bufB;
    for (int li = 0; li < 3; ++li) {
      int l = sc * 3 + li;
      launch_gemm(COUT[l] / 16, xin, Wh + l * 16384,
                  li == 0 ? idA : trA, li == 0 ? idS : trS, li == 0 ? 0 : 1,
                  xout, CINP[l], M, stream);
      hipMemsetAsync(gsum, 0, 512, stream);
      hipMemsetAsync(gsqs, 0, 512, stream);
      stats_kernel<<<(unsigned)((M + 255) / 256), 256, 0, stream>>>(
          xout, gsum, gsqs, COUT[l], M, 256);
      finalize_kernel<<<1, 128, 0, stream>>>(gsum, gsqs, gp[l], betap[l],
                                             trA, trS, 1.0f / (float)M, COUT[l]);
      _Float16* t = xin; xin = xout; xout = t;
    }
    int C3 = COUT[sc * 3 + 2];
    long te = (long)B_ * S_ * C3;
    maxk_kernel<<<(unsigned)((te + 255) / 256), 256, 0, stream>>>(
        xin, trA, trS, outFeat, K, C3, COFF[sc]);
  }
}
